// bin_sparseconnect_layer_9088150798855
// MI455X (gfx1250) — matrix operand order repaired and checked by reading, not compiled, not run on hardware
//
#include <hip/hip_runtime.h>

typedef _Float16 v16h __attribute__((ext_vector_type(16)));
typedef _Float16 v8h  __attribute__((ext_vector_type(8)));
typedef float    v8f  __attribute__((ext_vector_type(8)));
typedef int      v4i  __attribute__((ext_vector_type(4)));

#define UNITS 1024
#define D_IN  2048
#define KCONN 16
#define BATCH 4096
#define BK    32
#define LS    40   // LDS row stride in halfs (80 bytes: 16B-aligned, bank-spread)

#if __has_builtin(__builtin_amdgcn_global_load_async_to_lds_b128)
#define HAVE_ASYNC_LDS 1
#else
#define HAVE_ASYNC_LDS 0
#endif

#if __has_builtin(__builtin_amdgcn_s_wait_asynccnt)
#define WAIT_ASYNC() __builtin_amdgcn_s_wait_asynccnt(0)
#else
#define WAIT_ASYNC() asm volatile("s_wait_asynccnt 0x0" ::: "memory")
#endif

// ---------------------------------------------------------------------------
// Phase 1: per-unit top-16 of (D + GN), emit Bmask[u][k] = sign(W[u][k]) at
// the selected positions, 0 elsewhere. Row-major [UNITS][D_IN] f16 so GEMM
// B-tiles are contiguous along K.
// ---------------------------------------------------------------------------
__global__ __launch_bounds__(256) void topk_mask_kernel(
    const float* __restrict__ D, const float* __restrict__ GN,
    const float* __restrict__ W, _Float16* __restrict__ Bmask)
{
  __shared__ float pv[D_IN];
  __shared__ float rv[256];
  __shared__ int   ri[256];

  const int u = blockIdx.x;
  const int t = threadIdx.x;
  const float* Du = D  + (size_t)u * D_IN;
  const float* Gu = GN + (size_t)u * D_IN;
  _Float16*    Bu = Bmask + (size_t)u * D_IN;

  // zero this unit's mask row: 8 halfs (16B) per thread
  uint4 z; z.x = z.y = z.z = z.w = 0u;
  *(uint4*)&Bu[t * 8] = z;

  for (int i = t; i < D_IN; i += 256) pv[i] = Du[i] + Gu[i];
  __syncthreads();

  for (int it = 0; it < KCONN; ++it) {
    // local argmax (ascending i + strict > gives lowest-index tiebreak)
    float best = -__builtin_inff();
    int bidx = D_IN;
    for (int i = t; i < D_IN; i += 256) {
      float v = pv[i];
      if (v > best) { best = v; bidx = i; }
    }
    rv[t] = best; ri[t] = bidx;
    __syncthreads();
    for (int s = 128; s > 0; s >>= 1) {
      if (t < s) {
        float vo = rv[t + s]; int io = ri[t + s];
        if (vo > rv[t] || (vo == rv[t] && io < ri[t])) { rv[t] = vo; ri[t] = io; }
      }
      __syncthreads();
    }
    if (t == 0) {
      int j = ri[0];
      pv[j] = -__builtin_inff();
      float w = W[(size_t)u * D_IN + j];
      float s = (w > 0.f) ? 1.f : ((w < 0.f) ? -1.f : 0.f);
      Bu[j] = (_Float16)s;
    }
    __syncthreads();
  }
}

// ---------------------------------------------------------------------------
// Phase 2: y[4096,1024] = x[4096,2048] @ Bmask^T, f16 WMMA with f32 accum,
// hi/lo f16 split of x for ~f32 precision. Double-buffered LDS; B tile is
// copied with GLOBAL_LOAD_ASYNC_TO_LDS (ASYNCcnt) when available.
// ---------------------------------------------------------------------------
__device__ __forceinline__ v16h frag16(const _Float16* p0, const _Float16* p1)
{
  v8h a = *(const v8h*)p0;
  v8h b = *(const v8h*)p1;
  v16h r;
#pragma unroll
  for (int i = 0; i < 8; ++i) { r[i] = a[i]; r[i + 8] = b[i]; }
  return r;
}

union HalfPack4 { _Float16 h[4]; uint2 u; };

__global__ __launch_bounds__(256) void spgemm_wmma_kernel(
    const float* __restrict__ x, const _Float16* __restrict__ Bmask,
    float* __restrict__ y)
{
  __shared__ _Float16 Ah[2][128 * LS];
  __shared__ _Float16 Al[2][128 * LS];
  __shared__ _Float16 Bs[2][128 * LS];

  const int t     = threadIdx.x;
  const int lane  = t & 31;
  const int w     = t >> 5;
  const int wi    = w >> 2;       // 0..1 -> M half (64 rows)
  const int wj    = w & 3;        // 0..3 -> N quarter (32 cols)
  const int l15   = lane & 15;
  const int hi16  = lane >> 4;    // 0 or 1
  const int m_blk = blockIdx.y * 128;
  const int n_blk = blockIdx.x * 128;

  // --- staging helpers -----------------------------------------------------
  auto ldx = [&](int kc, float4* xr) {
#pragma unroll
    for (int i = 0; i < 4; ++i) {
      int idx = t + 256 * i, row = idx >> 3, c4 = idx & 7;
      xr[i] = *(const float4*)&x[(size_t)(m_blk + row) * D_IN + kc + c4 * 4];
    }
  };
  auto stx = [&](const float4* xr, int p) {
#pragma unroll
    for (int i = 0; i < 4; ++i) {
      int idx = t + 256 * i, row = idx >> 3, c4 = idx & 7;
      float f[4] = {xr[i].x, xr[i].y, xr[i].z, xr[i].w};
      HalfPack4 ph, pl;
#pragma unroll
      for (int c = 0; c < 4; ++c) {
        _Float16 h = (_Float16)f[c];
        ph.h[c] = h;
        pl.h[c] = (_Float16)(f[c] - (float)h);
      }
      *(uint2*)&Ah[p][row * LS + c4 * 4] = ph.u;
      *(uint2*)&Al[p][row * LS + c4 * 4] = pl.u;
    }
  };
#if HAVE_ASYNC_LDS
  auto stageB = [&](int kc, int p) {
#pragma unroll
    for (int i = 0; i < 2; ++i) {
      int idx = t + 256 * i, row = idx >> 2, seg = idx & 3;
      const _Float16* g = &Bmask[(size_t)(n_blk + row) * D_IN + kc + seg * 8];
      _Float16* l = &Bs[p][row * LS + seg * 8];
      __builtin_amdgcn_global_load_async_to_lds_b128(
          (__attribute__((address_space(1))) v4i*)g,
          (__attribute__((address_space(3))) v4i*)l, 0, 0);
    }
  };
#else
  auto ldb = [&](int kc, uint4* br) {
#pragma unroll
    for (int i = 0; i < 2; ++i) {
      int idx = t + 256 * i, row = idx >> 2, seg = idx & 3;
      br[i] = *(const uint4*)&Bmask[(size_t)(n_blk + row) * D_IN + kc + seg * 8];
    }
  };
  auto stb = [&](const uint4* br, int p) {
#pragma unroll
    for (int i = 0; i < 2; ++i) {
      int idx = t + 256 * i, row = idx >> 2, seg = idx & 3;
      *(uint4*)&Bs[p][row * LS + seg * 8] = br[i];
    }
  };
#endif

  v8f acc[4][2];
#pragma unroll
  for (int mi = 0; mi < 4; ++mi)
#pragma unroll
    for (int nj = 0; nj < 2; ++nj)
#pragma unroll
      for (int r = 0; r < 8; ++r) acc[mi][nj][r] = 0.f;

  // --- prologue: stage chunk 0 into buffer 0 -------------------------------
  {
    float4 xr[4];
    ldx(0, xr);
    stx(xr, 0);
#if HAVE_ASYNC_LDS
    stageB(0, 0);
    WAIT_ASYNC();
#else
    uint4 br[2];
    ldb(0, br);
    stb(br, 0);
#endif
  }
  __syncthreads();

  // --- main loop: compute buf p while staging buf p^1 ----------------------
  for (int it = 0; it < D_IN / BK; ++it) {
    const int p = it & 1;
    const int kc = it * BK;
    const bool more = (it + 1) < (D_IN / BK);

    float4 xr[4];
#if HAVE_ASYNC_LDS
    if (more) { stageB(kc + BK, p ^ 1); ldx(kc + BK, xr); }
#else
    uint4 br[2];
    if (more) { ldx(kc + BK, xr); ldb(kc + BK, br); }
#endif

    // B fragments: lanes 0-15 hold K{0..7,16..23} of column l15, 16-31 K{8..15,24..31}
    v16h bf[2];
#pragma unroll
    for (int nj = 0; nj < 2; ++nj) {
      int col = wj * 32 + nj * 16 + l15;
      int k0  = hi16 * 8;
      bf[nj] = frag16(&Bs[p][col * LS + k0], &Bs[p][col * LS + k0 + 16]);
    }
    // A fragments + WMMA: lanes 0-15 hold K{0..7,16..23}, 16-31 K{8..15,24..31}
#pragma unroll
    for (int mi = 0; mi < 4; ++mi) {
      int rowA = wi * 64 + mi * 16 + l15;
      int k0a  = hi16 * 8;
      v16h ah = frag16(&Ah[p][rowA * LS + k0a], &Ah[p][rowA * LS + k0a + 16]);
      v16h al = frag16(&Al[p][rowA * LS + k0a], &Al[p][rowA * LS + k0a + 16]);
#pragma unroll
      for (int nj = 0; nj < 2; ++nj) {
        acc[mi][nj] = __builtin_amdgcn_wmma_f32_16x16x32_f16(
            false, al, false, bf[nj], (short)0, acc[mi][nj], false, false);
        acc[mi][nj] = __builtin_amdgcn_wmma_f32_16x16x32_f16(
            false, ah, false, bf[nj], (short)0, acc[mi][nj], false, false);
      }
    }

    if (more) {
      stx(xr, p ^ 1);
#if !HAVE_ASYNC_LDS
      stb(br, p ^ 1);
#endif
#if HAVE_ASYNC_LDS
      WAIT_ASYNC();
#endif
    }
    __syncthreads();
  }

  // --- epilogue: C/D layout = VGPR r holds M=r (lanes 0-15) / M=8+r (16-31)
#pragma unroll
  for (int mi = 0; mi < 4; ++mi) {
    int mrow = m_blk + wi * 64 + mi * 16 + hi16 * 8;
#pragma unroll
    for (int nj = 0; nj < 2; ++nj) {
      int n = n_blk + wj * 32 + nj * 16 + l15;
#pragma unroll
      for (int r = 0; r < 8; ++r)
        y[(size_t)(mrow + r) * UNITS + n] = acc[mi][nj][r];
    }
  }
}

// ---------------------------------------------------------------------------
extern "C" void kernel_launch(void* const* d_in, const int* in_sizes, int n_in,
                              void* d_out, int out_size, void* d_ws, size_t ws_size,
                              hipStream_t stream)
{
  const float* x  = (const float*)d_in[0];   // [4096, 2048]
  const float* W  = (const float*)d_in[1];   // [1024, 2048]
  const float* D  = (const float*)d_in[2];   // [1024, 2048]
  const float* GN = (const float*)d_in[3];   // [1, 1024, 2048]
  // d_in[4] = k (fixed 16)
  _Float16* Bmask = (_Float16*)d_ws;         // [1024, 2048] f16 = 4 MB
  float* y = (float*)d_out;                  // [4096, 1024]

  topk_mask_kernel<<<UNITS, 256, 0, stream>>>(D, GN, W, Bmask);
  spgemm_wmma_kernel<<<dim3(UNITS / 128, BATCH / 128), 256, 0, stream>>>(x, Bmask, y);
}
